// SelfAttention_41480794145067
// MI455X (gfx1250) — matrix operand order repaired and checked by reading, not compiled, not run on hardware
//
#include <hip/hip_runtime.h>
#include <hip/hip_bf16.h>

// ---------------------------------------------------------------------------
// Problem constants (from the reference)
// ---------------------------------------------------------------------------
constexpr int kB  = 4;
constexpr int kS  = 2048;
constexpr int kE  = 1024;
constexpr int kH  = 16;
constexpr int kHD = 64;                    // head dim
constexpr float kScale = 0.125f;           // 1/sqrt(64), folded into Q

typedef __attribute__((ext_vector_type(16))) _Float16 v16h;
typedef __attribute__((ext_vector_type(8)))  _Float16 v8h;
typedef __attribute__((ext_vector_type(2)))  _Float16 v2h;
typedef __attribute__((ext_vector_type(8)))  float    v8f;
typedef __attribute__((ext_vector_type(8)))  unsigned v8u;

// ---------------------------------------------------------------------------
// WMMA fragment loaders (this code's contiguous k = 16h + s order, same for A and B, wave32)
//
// A (16x32, MxK, f16): lane L holds row M=L&15;
//   elements 0..15 = K = koff+0..15,
//   koff = 0 (lanes 0..15) or 16 (lanes 16..31).
// B (32x16, KxN, f16): lane L holds column N=L&15;
//   elements 0..15 = K = kb2+0..15, kb2 = 0 (lanes 0..15) or 16 (lanes 16..31).
//   B is loaded from a row-major [N][K] array ("transposed" storage), so every
//   GEMM here is X @ Y^T with fully contiguous per-lane loads.
// C/D (16x16, f32): lane L holds column N=L&15, element r = row r+(L<16?0:8).
// ---------------------------------------------------------------------------
__device__ __forceinline__ v16h load_a_half(const _Float16* __restrict__ tile,
                                            int lda, int lane) {
  const int m    = lane & 15;
  const int koff = (lane < 16) ? 0 : 16;
  const _Float16* p = tile + (size_t)m * lda + koff;
  v8h lo = *(const v8h*)p;
  v8h hi = *(const v8h*)(p + 8);
  v16h a;
#pragma unroll
  for (int i = 0; i < 8; ++i) { a[i] = lo[i]; a[i + 8] = hi[i]; }
  return a;
}

__device__ __forceinline__ v16h load_a_float(const float* __restrict__ tile,
                                             int lda, int lane) {
  const int m    = lane & 15;
  const int koff = (lane < 16) ? 0 : 16;
  const float* p = tile + (size_t)m * lda + koff;
  v16h a;
#pragma unroll
  for (int i = 0; i < 8; ++i) {
    a[i]     = (_Float16)p[i];
    a[i + 8] = (_Float16)p[8 + i];
  }
  return a;
}

__device__ __forceinline__ v16h load_b_half(const _Float16* __restrict__ tileT,
                                            int ldb, int lane) {
  const int n   = lane & 15;
  const int kb2 = (lane < 16) ? 0 : 16;
  return *(const v16h*)(tileT + (size_t)n * ldb + kb2);
}

__device__ __forceinline__ v8f wmma_f16(v16h a, v16h b, v8f c) {
  return __builtin_amdgcn_wmma_f32_16x16x32_f16(false, a, false, b,
                                                (short)0, c, false, false);
}

// lane <-> lane^16 half-swap as pure VALU (v_permlanex16_b32), no LDS pipe.
__device__ __forceinline__ unsigned xor16u(unsigned v) {
#if __has_builtin(__builtin_amdgcn_permlanex16)
  return (unsigned)__builtin_amdgcn_permlanex16(
      (int)v, (int)v, 0x76543210, 0xfedcba98, true, false);
#else
  return (unsigned)__shfl_xor((int)v, 16);
#endif
}
__device__ __forceinline__ float xor16f(float v) {
  return __builtin_bit_cast(float, xor16u(__builtin_bit_cast(unsigned, v)));
}

// ---------------------------------------------------------------------------
// f32 -> f16 weight conversion
// ---------------------------------------------------------------------------
__global__ __launch_bounds__(256) void cvt_f32_f16(const float* __restrict__ in,
                                                   _Float16* __restrict__ out,
                                                   int n) {
  int i = (blockIdx.x * 256 + threadIdx.x) * 4;
#pragma unroll
  for (int j = 0; j < 4; ++j)
    if (i + j < n) out[i + j] = (_Float16)in[i + j];
}

// ---------------------------------------------------------------------------
// GEMM: out = A[B*S, E] @ W^T + bias   (W is [E,E] row-major, f16)
// One wave computes a 16x64 tile of the output; 8 waves / block.
// MODE 0: Q proj  -> f16 [B,H,S,HD], scaled by 0.125
// MODE 1: K proj  -> f16 [B,H,S,HD]
// MODE 2: V proj  -> f16 [B,H,HD,S]  (transposed: V^T tiles are WMMA A-frags)
// MODE 3: O proj  -> f32 [B,S,E]     (A is f16 context)
// ---------------------------------------------------------------------------
template <int MODE>
__global__ __launch_bounds__(256) void gemm_proj(const void* __restrict__ Ap,
                                                 const _Float16* __restrict__ W,
                                                 const float* __restrict__ bias,
                                                 void* __restrict__ outp) {
  const int lane = threadIdx.x & 31;
  const int wid  = threadIdx.x >> 5;
  const int wave = blockIdx.x * 8 + wid;
  constexpr int ntiles = kE / 64;          // 16 n-tiles of width 64
  const int m0 = (wave / ntiles) * 16;
  const int n0 = (wave % ntiles) * 64;

  v8f acc[4] = {};
  for (int k0 = 0; k0 < kE; k0 += 32) {
    v16h a;
    if constexpr (MODE == 3)
      a = load_a_half((const _Float16*)Ap + (size_t)m0 * kE + k0, kE, lane);
    else
      a = load_a_float((const float*)Ap + (size_t)m0 * kE + k0, kE, lane);
#pragma unroll
    for (int t = 0; t < 4; ++t) {
      v16h b = load_b_half(W + (size_t)(n0 + 16 * t) * kE + k0, kE, lane);
      acc[t] = wmma_f16(a, b, acc[t]);
    }
  }

  const int off = (lane < 16) ? 0 : 8;
  const int nc  = lane & 15;
#pragma unroll
  for (int t = 0; t < 4; ++t) {
#pragma unroll
    for (int r = 0; r < 8; ++r) {
      const int sg = m0 + r + off;          // global row in [0, B*S)
      const int e  = n0 + 16 * t + nc;      // column in [0, E)
      float val = acc[t][r] + bias[e];
      const int bb = sg / kS, ss = sg % kS;
      const int hh = e / kHD, dd = e % kHD;
      if constexpr (MODE == 0) {
        ((_Float16*)outp)[(((size_t)bb * kH + hh) * kS + ss) * kHD + dd] =
            (_Float16)(val * kScale);
      } else if constexpr (MODE == 1) {
        ((_Float16*)outp)[(((size_t)bb * kH + hh) * kS + ss) * kHD + dd] =
            (_Float16)val;
      } else if constexpr (MODE == 2) {
        ((_Float16*)outp)[(((size_t)bb * kH + hh) * kHD + dd) * kS + ss] =
            (_Float16)val;
      } else {
        ((float*)outp)[(size_t)sg * kE + e] = val;
      }
    }
  }
}

// ---------------------------------------------------------------------------
// One 32-key flash-attention step, TRANSPOSED formulation:
//   S^T = K @ Q^T   (A = 16-key tiles of K, B = Q^T, C columns = queries)
//   O^T = V^T @ P^T (A = 16-HD tiles of Vt, B = P^T built via permlanex16)
// Each lane owns exactly one query column -> softmax stats are lane scalars;
// cross-lane traffic is 1 f32 half-swap (max) + 4 packed-f16 half-swaps (P).
// MASKED is true only for the single diagonal block per wave.
// ---------------------------------------------------------------------------
template <bool MASKED>
__device__ __forceinline__ void attn_step(
    const _Float16* __restrict__ Kh, const _Float16* __restrict__ Vth,
    const v16h& qb0, const v16h& qb1,
    v8f (&o)[4], float& rmax, float& rsum,
    int kb, int q0, int lane, int off, int nc) {
  const bool lowHalf = (lane < 16);

  // ---- S^T tiles: st0 = keys kb..kb+15, st1 = keys kb+16..kb+31 ----
  v8f st0 = {}, st1 = {};
  {
    v16h ka00 = load_a_half(Kh + (size_t)(kb +  0) * kHD + 0,  kHD, lane);
    v16h ka10 = load_a_half(Kh + (size_t)(kb + 16) * kHD + 0,  kHD, lane);
    st0 = wmma_f16(ka00, qb0, st0);
    st1 = wmma_f16(ka10, qb0, st1);
    v16h ka01 = load_a_half(Kh + (size_t)(kb +  0) * kHD + 32, kHD, lane);
    v16h ka11 = load_a_half(Kh + (size_t)(kb + 16) * kHD + 32, kHD, lane);
    st0 = wmma_f16(ka01, qb1, st0);
    st1 = wmma_f16(ka11, qb1, st1);
  }

  // ---- (optional causal mask) + lane-local row max ----
  const int qcol = q0 + nc;                  // this lane's query
  float e0[8], e1[8];
#pragma unroll
  for (int r = 0; r < 8; ++r) {
    e0[r] = st0[r];                          // key = kb + r + off
    e1[r] = st1[r];                          // key = kb + 16 + r + off
    if constexpr (MASKED) {
      const int key = kb + r + off;
      if (key > qcol)      e0[r] = -1.0e30f;
      if (key + 16 > qcol) e1[r] = -1.0e30f;
    }
  }
  float mloc = fmaxf(e0[0], e1[0]);
#pragma unroll
  for (int r = 1; r < 8; ++r) mloc = fmaxf(mloc, fmaxf(e0[r], e1[r]));
  const float mx   = fmaxf(mloc, xor16f(mloc));          // partner half
  const float mnew = fmaxf(rmax, mx);
  const float corr = __expf(rmax - mnew);
  rmax = mnew;

  // ---- exponentiate, accumulate partial row sum, rescale O ----
  float p0[8], p1[8], psum = 0.0f;
#pragma unroll
  for (int r = 0; r < 8; ++r) {
    p0[r] = __expf(e0[r] - mnew);
    p1[r] = __expf(e1[r] - mnew);
    psum += p0[r] + p1[r];
  }
  rsum = rsum * corr + psum;
#pragma unroll
  for (int c = 0; c < 4; ++c)
#pragma unroll
    for (int r = 0; r < 8; ++r) o[c][r] *= corr;

  // ---- build P^T B-fragment with packed-f16 half-swaps ----
  // B element j <-> key kb2+j. Dword j of the fragment = elements 2j,2j+1.
  // lane<16 covers keys 0..15: dwords 0..3 own tile0, dwords 4..7 partner tile0
  // lane>=16 covers keys 16..31: dwords 0..3 partner tile1, dwords 4..7 own t1
  unsigned pk0[4], pk1[4];
#pragma unroll
  for (int j = 0; j < 4; ++j) {
    pk0[j] = __builtin_bit_cast(
        unsigned, __builtin_amdgcn_cvt_pkrtz(p0[2 * j], p0[2 * j + 1]));
    pk1[j] = __builtin_bit_cast(
        unsigned, __builtin_amdgcn_cvt_pkrtz(p1[2 * j], p1[2 * j + 1]));
  }
  v8u pbu;
#pragma unroll
  for (int j = 0; j < 4; ++j) {
    const unsigned send = lowHalf ? pk1[j] : pk0[j];   // what partner needs
    const unsigned recv = xor16u(send);                // partner's send
    pbu[j]     = lowHalf ? pk0[j] : recv;              // keys kb2+2j..2j+1
    pbu[4 + j] = lowHalf ? recv : pk1[j];              // keys kb2+8+2j..
  }
  const v16h pb = __builtin_bit_cast(v16h, pbu);

  // ---- O^T += V^T @ P^T  (Vt rows are HD dims -> contiguous A fragments) ----
#pragma unroll
  for (int c = 0; c < 4; ++c) {
    v16h va = load_a_half(Vth + (size_t)(16 * c) * kS + kb, kS, lane);
    o[c] = wmma_f16(va, pb, o[c]);
  }
}

// ---------------------------------------------------------------------------
// Flash attention (causal), f16 operands / f32 accum, transposed tiles.
// One wave owns 16 query columns; 8 waves per block -> 128 queries per block.
// Main loop runs mask-free; exactly one masked diagonal step finishes.
// ---------------------------------------------------------------------------
__global__ __launch_bounds__(256) void attn_flash(
    const _Float16* __restrict__ Q,   // [B,H,S,HD], pre-scaled
    const _Float16* __restrict__ K,   // [B,H,S,HD]
    const _Float16* __restrict__ Vt,  // [B,H,HD,S]
    _Float16* __restrict__ ctx)       // [B,S,E]
{
  const int lane = threadIdx.x & 31;
  const int wid  = threadIdx.x >> 5;
  constexpr int qblocks = kS / 128;
  const int bh   = blockIdx.x / qblocks;
  const int qblk = blockIdx.x % qblocks;
  const int bb   = bh / kH;
  const int q0   = qblk * 128 + wid * 16;

  const _Float16* Qh  = Q  + (size_t)bh * kS * kHD;
  const _Float16* Kh  = K  + (size_t)bh * kS * kHD;
  const _Float16* Vth = Vt + (size_t)bh * kHD * kS;

  const int off = (lane < 16) ? 0 : 8;
  const int nc  = lane & 15;

  // Q^T B-fragments for HD = 0..31 and 32..63 (loop-invariant)
  const v16h qb0 = load_b_half(Qh + (size_t)q0 * kHD + 0,  kHD, lane);
  const v16h qb1 = load_b_half(Qh + (size_t)q0 * kHD + 32, kHD, lane);

  v8f o[4] = {};
  float rmax = -3.0e38f, rsum = 0.0f;

  // Blocks [0, kdiag) are fully below the diagonal for every query this wave
  // owns (queries q0..q0+15); the block at kdiag is the only partially-masked
  // one, and no block is ever fully masked (q0 = 16a, kdiag = q0 & ~31).
  const int kdiag = q0 & ~31;
  const int m     = lane & 15;
  for (int kb = 0; kb < kdiag; kb += 32) {
    // Prefetch the next key block (global_prefetch_b8); guarded so the
    // non-speculative WGP prefetch never walks past this head's sequence.
    if (kb + 64 <= kS) {
      const int pn = kb + 32;
#pragma unroll
      for (int t = 0; t < 2; ++t) {
        __builtin_prefetch(Kh + (size_t)(pn + 16 * t + m) * kHD + 0,  0, 1);
        __builtin_prefetch(Kh + (size_t)(pn + 16 * t + m) * kHD + 32, 0, 1);
      }
#pragma unroll
      for (int c = 0; c < 4; ++c)
        __builtin_prefetch(Vth + (size_t)(16 * c + m) * kS + pn, 0, 1);
    }
    attn_step<false>(Kh, Vth, qb0, qb1, o, rmax, rsum, kb, q0, lane, off, nc);
  }
  attn_step<true>(Kh, Vth, qb0, qb1, o, rmax, rsum, kdiag, q0, lane, off, nc);

  // ---- normalize; each lane stores 8 contiguous f16 per c-tile ----
  const float rtot = rsum + xor16f(rsum);
  const float inv  = 1.0f / rtot;
  const int hh = bh % kH;
  _Float16* crow = ctx + ((size_t)bb * kS + q0 + nc) * kE + hh * kHD + off;
#pragma unroll
  for (int c = 0; c < 4; ++c) {
    v8h outv;
#pragma unroll
    for (int r = 0; r < 8; ++r) outv[r] = (_Float16)(o[c][r] * inv);
    *(v8h*)(crow + 16 * c) = outv;
  }
}

// ---------------------------------------------------------------------------
// Launcher
// ---------------------------------------------------------------------------
extern "C" void kernel_launch(void* const* d_in, const int* in_sizes, int n_in,
                              void* d_out, int out_size, void* d_ws,
                              size_t ws_size, hipStream_t stream) {
  (void)in_sizes; (void)n_in; (void)out_size; (void)ws_size;
  const float* values  = (const float*)d_in[0];
  const float* keys    = (const float*)d_in[1];
  const float* queries = (const float*)d_in[2];
  const float* Wv = (const float*)d_in[4];  const float* bv = (const float*)d_in[5];
  const float* Wk = (const float*)d_in[6];  const float* bk = (const float*)d_in[7];
  const float* Wq = (const float*)d_in[8];  const float* bq = (const float*)d_in[9];
  const float* Wo = (const float*)d_in[10]; const float* bo = (const float*)d_in[11];

  // workspace layout (f16): 4 weights (E*E) + Q + K + Vt + ctx (B*S*E each)
  _Float16* ws  = (_Float16*)d_ws;
  const size_t wsz = (size_t)kE * kE;
  const size_t asz = (size_t)kB * kS * kE;
  _Float16* wqh = ws;
  _Float16* wkh = wqh + wsz;
  _Float16* wvh = wkh + wsz;
  _Float16* woh = wvh + wsz;
  _Float16* qh  = woh + wsz;
  _Float16* kh  = qh + asz;
  _Float16* vt  = kh + asz;
  _Float16* ctx = vt + asz;

  const int nW = kE * kE;
  const int cvtBlocks = nW / 1024;
  cvt_f32_f16<<<cvtBlocks, 256, 0, stream>>>(Wq, wqh, nW);
  cvt_f32_f16<<<cvtBlocks, 256, 0, stream>>>(Wk, wkh, nW);
  cvt_f32_f16<<<cvtBlocks, 256, 0, stream>>>(Wv, wvh, nW);
  cvt_f32_f16<<<cvtBlocks, 256, 0, stream>>>(Wo, woh, nW);

  // (B*S/16) m-tiles * (E/64) n-tiles = 8192 waves -> 1024 blocks of 8 waves
  const int gemmBlocks = ((kB * kS / 16) * (kE / 64)) / 8;
  gemm_proj<0><<<gemmBlocks, 256, 0, stream>>>(queries, wqh, bq, qh);
  gemm_proj<1><<<gemmBlocks, 256, 0, stream>>>(keys,    wkh, bk, kh);
  gemm_proj<2><<<gemmBlocks, 256, 0, stream>>>(values,  wvh, bv, vt);

  attn_flash<<<kB * kH * (kS / 128), 256, 0, stream>>>(qh, kh, vt, ctx);

  gemm_proj<3><<<gemmBlocks, 256, 0, stream>>>(ctx, woh, bo, d_out);
}
